// GATLayer_46892452937743
// MI455X (gfx1250) — matrix operand order repaired and checked by reading, not compiled, not run on hardware
//
#include <hip/hip_runtime.h>
#include <hip/hip_bf16.h>

// ---------------------------------------------------------------------------
// GAT layer for MI455X (gfx1250): flash-style masked softmax + WMMA bf16
// + TDM (tensor_load_to_lds) double-buffered Z staging.
// B=4, N=2048, IN_DIM=256, H=4, D=64, TOTAL=256
// ---------------------------------------------------------------------------

#define GAT_B     4
#define GAT_N     2048
#define GAT_K     256     // IN_DIM
#define GAT_H     4
#define GAT_D     64
#define GAT_TOT   256     // H*D
#define LDSTR     40      // LDS row stride in bf16 elems: 80B, 16B aligned

typedef __bf16 bf16_t;
typedef __attribute__((ext_vector_type(16))) __bf16 v16bf;
typedef __attribute__((ext_vector_type(8)))  __bf16 v8bf;
typedef __attribute__((ext_vector_type(8)))  float  v8f;
typedef __attribute__((ext_vector_type(4)))  unsigned int u32x4;
typedef __attribute__((ext_vector_type(8)))  int i32x8;
typedef __attribute__((ext_vector_type(4)))  int i32x4;

// Assemble a 16-element bf16 WMMA operand from two 16-byte LDS chunks.
__device__ __forceinline__ v16bf mk16(const bf16_t* p0, const bf16_t* p1) {
    v8bf lo = *(const v8bf*)p0;
    v8bf hi = *(const v8bf*)p1;
    v16bf r;
#pragma unroll
    for (int i = 0; i < 8; ++i) { r[i] = lo[i]; r[i + 8] = hi[i]; }
    return r;
}

__device__ __forceinline__ v8f wmma_bf16(v16bf a, v16bf b, v8f c) {
    return __builtin_amdgcn_wmma_f32_16x16x32_bf16(
        /*neg_a=*/false, a, /*neg_b=*/false, b,
        /*c_mod=*/(short)0, c, /*reuse_a=*/false, /*reuse_b=*/false);
}

// ---------------------------------------------------------------------------
// TDM: 2D tile load Global -> LDS with per-row LDS padding.
// D# packed per CDNA5 ISA §8.3/§8.4. data_size = 2B (bf16).
// pad_interval=3 (pad every 16 DWORDs = 64B row), pad_amount=3 (4 DWORDs = 16B)
// -> effective LDS row stride 80B == LDSTR bf16 elems.
// ---------------------------------------------------------------------------
__device__ __forceinline__ void tdm_load_tile_bf16(
        unsigned lds_off, const bf16_t* gsrc,
        unsigned tile_x, unsigned tile_y,
        unsigned tensor_x, unsigned tensor_y, unsigned stride_elems) {
    unsigned long long ga = (unsigned long long)(uintptr_t)gsrc;
    u32x4 g0;
    g0[0] = 1u;                                   // count=1 (valid descriptor)
    g0[1] = lds_off;                              // lds_addr
    g0[2] = (unsigned)(ga & 0xFFFFFFFFu);         // global_addr[31:0]
    g0[3] = (unsigned)((ga >> 32) & 0x01FFFFFFu)  // global_addr[56:32]
          | (2u << 30);                           // type=2 ("image")

    i32x8 g1;
    g1[0] = (int)((1u << 16)      // data_size = 1 -> 2 bytes
                | (1u << 20)      // pad_enable
                | (3u << 22)      // pad_interval: 16 DWORDs
                | (3u << 25));    // pad_amount:   4 DWORDs
    g1[1] = (int)((tensor_x & 0xFFFFu) << 16);                        // dim0[15:0]
    g1[2] = (int)(((tensor_x >> 16) & 0xFFFFu) |
                  ((tensor_y & 0xFFFFu) << 16));                      // dim0[31:16]|dim1[15:0]
    g1[3] = (int)(((tensor_y >> 16) & 0xFFFFu) |
                  ((tile_x & 0xFFFFu) << 16));                        // dim1[31:16]|tile0
    g1[4] = (int)(tile_y & 0xFFFFu);                                  // tile1, tile2=0
    g1[5] = (int)stride_elems;                                        // dim0_stride[31:0]
    g1[6] = 0;                                                        // dim0_stride[47:32]
    g1[7] = 0;                                                        // dim1_stride
    i32x4 z4 = {0, 0, 0, 0};
    i32x8 z8 = {0, 0, 0, 0, 0, 0, 0, 0};
    __builtin_amdgcn_tensor_load_to_lds(g0, g1, z4, z4, z8, 0);
}

// ---------------------------------------------------------------------------
// Kernel 1: projected = nodes @ W_proj + b_proj  ([8192,256]x[256,256])
// Also emits zt[b][h][c][j] bf16 (transposed, TDM-ready) from the epilogue.
// ---------------------------------------------------------------------------
__global__ void __launch_bounds__(256)
proj_gemm(const float* __restrict__ nodes, const float* __restrict__ W,
          const float* __restrict__ bias, float* __restrict__ proj,
          bf16_t* __restrict__ zt) {
    const int row0 = blockIdx.x * 128;
    const int n0   = blockIdx.y * 16;
    const int tid  = threadIdx.x;
    const int w    = tid >> 5;
    const int l    = tid & 31;

    __shared__ alignas(16) bf16_t As[128 * LDSTR];   // A tile, row-major (M x K)
    __shared__ alignas(16) bf16_t Bs[16 * LDSTR];    // B tile, column-major (N x K)

    v8f acc = {0.f, 0.f, 0.f, 0.f, 0.f, 0.f, 0.f, 0.f};

    for (int k0 = 0; k0 < GAT_K; k0 += 32) {
#pragma unroll 4
        for (int idx = tid; idx < 128 * 32; idx += 256) {
            int r = idx >> 5, kk = idx & 31;
            As[r * LDSTR + kk] = (bf16_t)nodes[(size_t)(row0 + r) * GAT_K + k0 + kk];
        }
#pragma unroll
        for (int idx = tid; idx < 32 * 16; idx += 256) {
            int kk = idx >> 4, nn = idx & 15;
            Bs[nn * LDSTR + kk] = (bf16_t)W[(size_t)(k0 + kk) * GAT_TOT + n0 + nn];
        }
        __syncthreads();

        const int rl = w * 16 + (l & 15);
        const int ka = (l >> 4) * 8;
        v16bf a = mk16(&As[rl * LDSTR + ka], &As[rl * LDSTR + ka + 16]);
        const int cb = (l & 15);
        const int kb = (l >> 4) * 8;
        v16bf b = mk16(&Bs[cb * LDSTR + kb], &Bs[cb * LDSTR + kb + 16]);

        acc = wmma_bf16(a, b, acc);
        __syncthreads();
    }

    // C/D layout: VGPR i -> M = i (lanes 0-15) / i+8 (lanes 16-31), N = l%16
    const int col   = n0 + (l & 15);
    const float bv  = bias[col];
    const int rbase = row0 + w * 16 + ((l >> 4) ? 8 : 0);
    v8bf zp;
#pragma unroll
    for (int i = 0; i < 8; ++i) {
        float v = acc[i] + bv;
        proj[(size_t)(rbase + i) * GAT_TOT + col] = v;
        zp[i] = (bf16_t)v;
    }
    // 8 consecutive rows (same col) == 8 consecutive j in zt -> one 16B store
    const int bb = rbase / GAT_N;
    const int jb = rbase % GAT_N;
    const int hh = col / GAT_D;
    const int cc = col % GAT_D;
    *(v8bf*)(zt + (((size_t)(bb * GAT_H + hh) * GAT_D + cc) * GAT_N + jb)) = zp;
}

// ---------------------------------------------------------------------------
// Kernel 2: src/tgt = projected @ a_src / a_tgt   (one wave per row)
// ---------------------------------------------------------------------------
__global__ void __launch_bounds__(256)
attn_vec(const float* __restrict__ proj, const float* __restrict__ a_src,
         const float* __restrict__ a_tgt, float* __restrict__ src,
         float* __restrict__ tgt) {
    const int w   = threadIdx.x >> 5;
    const int l   = threadIdx.x & 31;
    const int row = blockIdx.x * 8 + w;

    const float4* as4 = (const float4*)a_src;   // [256][4]
    const float4* at4 = (const float4*)a_tgt;
    const float*  pr  = proj + (size_t)row * GAT_TOT;

    float s0 = 0, s1 = 0, s2 = 0, s3 = 0, t0 = 0, t1 = 0, t2 = 0, t3 = 0;
#pragma unroll
    for (int c = l; c < GAT_TOT; c += 32) {
        float  pv = pr[c];
        float4 a  = as4[c];
        float4 t  = at4[c];
        s0 += pv * a.x; s1 += pv * a.y; s2 += pv * a.z; s3 += pv * a.w;
        t0 += pv * t.x; t1 += pv * t.y; t2 += pv * t.z; t3 += pv * t.w;
    }
#pragma unroll
    for (int off = 16; off > 0; off >>= 1) {
        s0 += __shfl_down(s0, off); s1 += __shfl_down(s1, off);
        s2 += __shfl_down(s2, off); s3 += __shfl_down(s3, off);
        t0 += __shfl_down(t0, off); t1 += __shfl_down(t1, off);
        t2 += __shfl_down(t2, off); t3 += __shfl_down(t3, off);
    }
    if (l == 0) {
        ((float4*)src)[row] = make_float4(s0, s1, s2, s3);
        ((float4*)tgt)[row] = make_float4(t0, t1, t2, t3);
    }
}

// ---------------------------------------------------------------------------
// Kernel 3: masked softmax (flash-style) + WMMA aggregation.
// Z tiles arrive via TDM double-buffered async copies (TENSORcnt pipeline).
// Block: 256 thr handles (b, h, 64-row i-tile) -> out tile 64x64.
// ---------------------------------------------------------------------------
__global__ void __launch_bounds__(256)
gat_attn(const bf16_t* __restrict__ zt, const int* __restrict__ adj,
         const float* __restrict__ src, const float* __restrict__ tgt,
         float* __restrict__ out) {
    const int i0  = blockIdx.x * 64;
    const int h   = blockIdx.y;
    const int b   = blockIdx.z;
    const int tid = threadIdx.x;
    const int w   = tid >> 5;
    const int l   = tid & 31;

    __shared__ float tgt_s[GAT_N];
    __shared__ float src_s[64];
    __shared__ float sm[64];
    __shared__ float sd[64];
    __shared__ alignas(16) bf16_t Pt[64 * LDSTR];       // P tile 64 x 32 (bf16)
    __shared__ alignas(16) bf16_t Zb[2][64 * LDSTR];    // Z^T tiles 64c x 32j

    const bf16_t* zhead = zt + (size_t)(b * GAT_H + h) * GAT_D * GAT_N;

    // Kick off TDM for the first two Z tiles; they overlap with pass 1.
    if (w == 0) {
        tdm_load_tile_bf16((unsigned)(uintptr_t)&Zb[0][0], zhead + 0,
                           32, 64, GAT_N, GAT_D, GAT_N);
        tdm_load_tile_bf16((unsigned)(uintptr_t)&Zb[1][0], zhead + 32,
                           32, 64, GAT_N, GAT_D, GAT_N);
    }

    // Stage per-(b,h) logit vectors
    for (int idx = tid; idx < GAT_N; idx += 256)
        tgt_s[idx] = tgt[((size_t)b * GAT_N + idx) * GAT_H + h];
    if (tid < 64)
        src_s[tid] = src[((size_t)b * GAT_N + i0 + tid) * GAT_H + h];
    __syncthreads();

    const int* adjb = adj + (size_t)b * GAT_N * GAT_N;

    // ---- Pass 1: per-row online (max, sum) over masked leaky logits ----
    for (int rr = 0; rr < 8; ++rr) {
        const int   r    = w * 8 + rr;
        const int*  arow = adjb + (size_t)(i0 + r) * GAT_N;
        const float sv   = src_s[r];
        float m = -1e30f, s = 0.0f;
        for (int j = l; j < GAT_N; j += 32) {
            float e = sv + tgt_s[j];
            e = (e >= 0.0f) ? e : 0.2f * e;
            if (arow[j] > 0) {
                if (e > m) { s = s * __expf(m - e) + 1.0f; m = e; }
                else       { s += __expf(e - m); }
            }
        }
#pragma unroll
        for (int off = 16; off > 0; off >>= 1) {
            float mo = __shfl_down(m, off);
            float so = __shfl_down(s, off);
            float M  = fmaxf(m, mo);
            s = s * __expf(m - M) + so * __expf(mo - M);
            m = M;
        }
        if (l == 0) { sm[r] = (m < -1e29f) ? 0.0f : m; sd[r] = s; }
    }
    __syncthreads();

    // ---- Pass 2: out[64x64] += P(64x32) @ Z(32x64) over j tiles ----
    const int tr  = (w & 3) * 16;
    const int tc0 = (w >> 2) * 32;
    v8f acc0 = {0.f, 0.f, 0.f, 0.f, 0.f, 0.f, 0.f, 0.f};
    v8f acc1 = {0.f, 0.f, 0.f, 0.f, 0.f, 0.f, 0.f, 0.f};

    const int   pr_ = tid >> 2;           // P-stage row      0..63
    const int   pj0 = (tid & 3) * 8;      // P-stage jj base  {0,8,16,24}
    const float pm  = sm[pr_];
    const float psv = src_s[pr_];

    for (int t = 0; t < GAT_N / 32; ++t) {
        const int j0 = t * 32;
        // Stage P: p = mask * exp(leaky(src_i + tgt_j) - m_i), bf16
        const int* arow = adjb + (size_t)(i0 + pr_) * GAT_N + j0 + pj0;
        if (j0 + 32 < GAT_N) __builtin_prefetch(arow + 32, 0, 1);
#pragma unroll
        for (int e8 = 0; e8 < 8; ++e8) {
            float e = psv + tgt_s[j0 + pj0 + e8];
            e = (e >= 0.0f) ? e : 0.2f * e;
            float p = (arow[e8] > 0) ? __expf(e - pm) : 0.0f;
            Pt[pr_ * LDSTR + pj0 + e8] = (bf16_t)p;
        }
        // Z tile t must be resident (tiles complete in issue order).
        if (w == 0) {
            if (j0 + 32 < GAT_N) __builtin_amdgcn_s_wait_tensorcnt((short)1);
            else                 __builtin_amdgcn_s_wait_tensorcnt((short)0);
        }
        __syncthreads();

        const bf16_t* Zc = &Zb[t & 1][0];
        const int rl = tr + (l & 15);
        const int ka = (l >> 4) * 8;
        v16bf a  = mk16(&Pt[rl * LDSTR + ka], &Pt[rl * LDSTR + ka + 16]);
        const int kb = (l >> 4) * 8;
        const int c0 = tc0 + (l & 15);
        v16bf b0 = mk16(&Zc[c0 * LDSTR + kb], &Zc[c0 * LDSTR + kb + 16]);
        v16bf b1 = mk16(&Zc[(c0 + 16) * LDSTR + kb], &Zc[(c0 + 16) * LDSTR + kb + 16]);
        acc0 = wmma_bf16(a, b0, acc0);
        acc1 = wmma_bf16(a, b1, acc1);
        __syncthreads();

        // Refill the buffer just consumed with tile t+2.
        if (w == 0 && t + 2 < GAT_N / 32) {
            tdm_load_tile_bf16((unsigned)(uintptr_t)&Zb[t & 1][0],
                               zhead + (size_t)(t + 2) * 32,
                               32, 64, GAT_N, GAT_D, GAT_N);
        }
    }

    // Epilogue: divide by softmax denominator (0 if row fully masked), store
    const int rloc = tr + ((l >> 4) ? 8 : 0);
    const int cc   = l & 15;
#pragma unroll
    for (int i = 0; i < 8; ++i) {
        const int   rrow = rloc + i;
        const float d    = sd[rrow];
        const float inv  = (d > 0.0f) ? 1.0f / d : 0.0f;
        const size_t o = ((size_t)b * GAT_N + i0 + rrow) * GAT_TOT + h * GAT_D;
        out[o + tc0 + cc]      = acc0[i] * inv;
        out[o + tc0 + 16 + cc] = acc1[i] * inv;
    }
}

// ---------------------------------------------------------------------------
extern "C" void kernel_launch(void* const* d_in, const int* in_sizes, int n_in,
                              void* d_out, int out_size, void* d_ws, size_t ws_size,
                              hipStream_t stream) {
    const float* nodes     = (const float*)d_in[0];
    const int*   adjacency = (const int*)d_in[1];
    const float* W_proj    = (const float*)d_in[2];
    const float* b_proj    = (const float*)d_in[3];
    const float* a_src     = (const float*)d_in[4];
    const float* a_tgt     = (const float*)d_in[5];
    float*       out       = (float*)d_out;

    char*   ws   = (char*)d_ws;
    float*  proj = (float*)ws;                                         // 8 MB
    bf16_t* zt   = (bf16_t*)(ws + (size_t)8 * 1024 * 1024);            // 4 MB
    float*  src  = (float*)(ws + (size_t)12 * 1024 * 1024);            // 128 KB
    float*  tgt  = (float*)(ws + (size_t)12 * 1024 * 1024 + 131072);   // 128 KB

    proj_gemm<<<dim3(64, 16), 256, 0, stream>>>(nodes, W_proj, b_proj, proj, zt);
    attn_vec<<<dim3(1024), 256, 0, stream>>>(proj, a_src, a_tgt, src, tgt);
    gat_attn<<<dim3(GAT_N / 64, GAT_H, GAT_B), 256, 0, stream>>>(zt, adjacency,
                                                                 src, tgt, out);
}
